// BernoulliIndependentGenerator_14508399526051
// MI455X (gfx1250) — compile-verified, matrix operand order checked by reading, not run on hardware
//
#include <hip/hip_runtime.h>
#include <hip/hip_bf16.h>

// ---------------------------------------------------------------------------
// BiLSTM rationale-extraction (BernoulliGate top-k) for MI455X / gfx1250.
//   B=16, S=1024, E=256, H=256, 4H=1024, V=50257, BUDGET=10%
// Pipeline:
//   k_embed   : gather embed_table[x] -> f16            [B*S, E]
//   k_cvt     : w_ih_{f,b}, w_hh_{f,b} f32 -> f16 (row-major [1024,256])
//   k_inproj  : xp[d] = emb @ w_ih_d^T + b_d  via v_wmma_f32_16x16x32_f16
//   k_lstm    : 2 workgroups (fwd/bwd), 16 waves each; h double-buffered in
//               LDS (f16); 32 WMMAs / wave / step; masked packed-seq update
//   k_gate    : probs = sigmoid([h_f,h_b]·z_w + z_b), masked
//   k_topk    : per-row rank counting -> z in {0,1}
// ---------------------------------------------------------------------------

typedef __attribute__((ext_vector_type(16))) _Float16 v16h;
typedef __attribute__((ext_vector_type(8)))  _Float16 v8h;
typedef __attribute__((ext_vector_type(8)))  float    v8f;

#define Bsz 16
#define Ssz 1024
#define Esz 256
#define Hsz 256
#define G4  1024   // 4*H
#define HP  264    // padded LDS row stride (f16 elems) to dodge bank conflicts

__device__ __forceinline__ float sigmoidf_(float x) {
  return 1.0f / (1.0f + __expf(-x));
}

__device__ __forceinline__ v8f wmma_f16(v16h a, v16h b, v8f c) {
  // (neg_a, A, neg_b, B, c_mod, C, reuse_a, reuse_b)
  return __builtin_amdgcn_wmma_f32_16x16x32_f16(false, a, false, b,
                                                (short)0, c, false, false);
}

// Load a 16x32 A/B fragment from a row-major f16 matrix (tile origin at p,
// leading dim ld elems).  Matches the gfx1250 16-bit A layout:
//   lane: row = lane&15, k-base = (lane>>4)*8; VGPR0-3 = k[kb..kb+7],
//   VGPR4-7 = k[16+kb..16+kb+7].  B fragment == A fragment of W (W is [N,K]).
__device__ __forceinline__ v16h load_frag(const _Float16* __restrict__ p, int ld) {
  int lane = threadIdx.x & 31;
  int m    = lane & 15;
  int kb   = (lane >> 4) << 3;
  const _Float16* r = p + m * ld + kb;
  v8h lo = *(const v8h*)(r);
  v8h hi = *(const v8h*)(r + 16);
  v16h out;
#pragma unroll
  for (int i = 0; i < 8; ++i) { out[i] = lo[i]; out[i + 8] = hi[i]; }
  return out;
}

// ---------------------------------------------------------------- k_embed ---
__global__ __launch_bounds__(256) void k_embed(const int* __restrict__ x,
                                               const float* __restrict__ tab,
                                               _Float16* __restrict__ emb) {
  int i = blockIdx.x * blockDim.x + threadIdx.x;   // over B*S*E
  if (i >= Bsz * Ssz * Esz) return;
  int row = i >> 8, e = i & 255;
  emb[i] = (_Float16)tab[x[row] * Esz + e];
}

// ------------------------------------------------------------------ k_cvt ---
__global__ __launch_bounds__(256) void k_cvt(const float* __restrict__ src,
                                             _Float16* __restrict__ dst, int n) {
  int i = blockIdx.x * blockDim.x + threadIdx.x;
  if (i < n) dst[i] = (_Float16)src[i];
}

// --------------------------------------------------------------- k_inproj ---
// xp[d][m][n] = sum_k emb[m][k] * w_ih_d[n][k] + bias_d[n]
// One wave per 16x16 output tile; K = 256 -> 8 WMMA per tile.
__global__ __launch_bounds__(256) void k_inproj(const _Float16* __restrict__ emb,
                                                const _Float16* __restrict__ w_ih, // [2][1024][256]
                                                const float* __restrict__ bf,
                                                const float* __restrict__ bb,
                                                float* __restrict__ xp) {          // [2][16384][1024]
  int wid  = (blockIdx.x * blockDim.x + threadIdx.x) >> 5;   // 0..131071
  int lane = threadIdx.x & 31;
  int dir  = wid >> 16;
  int rem  = wid & 65535;
  int mt   = rem >> 6;          // 0..1023  (16 rows of B*S each)
  int nt   = rem & 63;          // 0..63    (16 cols of 4H each)
  int n0   = nt << 4;

  const _Float16* Wd  = w_ih + dir * (G4 * Esz);
  const float*    bia = dir ? bb : bf;

  float bv = bia[n0 + (lane & 15)];
  v8f acc;
#pragma unroll
  for (int r = 0; r < 8; ++r) acc[r] = bv;

  const _Float16* At = emb + (mt << 4) * Esz;
  const _Float16* Bt = Wd + n0 * Esz;
#pragma unroll
  for (int kk = 0; kk < 8; ++kk) {
    v16h a = load_frag(At + (kk << 5), Esz);
    v16h b = load_frag(Bt + (kk << 5), Esz);
    acc = wmma_f16(a, b, acc);
  }

  float* out = xp + (size_t)dir * (Bsz * Ssz * G4) + (size_t)(mt << 4) * G4 + n0;
  int half = lane >> 4, n = lane & 15;
#pragma unroll
  for (int r = 0; r < 8; ++r) {
    int m = r + (half << 3);
    out[(size_t)m * G4 + n] = acc[r];
  }
}

// ----------------------------------------------------------------- k_lstm ---
// blockIdx.x = direction (0 fwd, 1 bwd). 512 threads = 16 waves.
// Wave w owns hidden columns j = 16w..16w+15 across all 4 gates
// (gate-tile column blocks 256*g + 16w). h is double-buffered f16 in LDS.
__global__ __launch_bounds__(512) void k_lstm(const float* __restrict__ xp,      // [2][B*S][4H]
                                              const _Float16* __restrict__ w_hh, // [2][1024][256]
                                              const unsigned char* __restrict__ mask, // [B][S] (bool)
                                              float* __restrict__ h_out) {       // [2][B*S][H]
  int dir = blockIdx.x;
  const float*    xpd = xp   + (size_t)dir * (Bsz * Ssz * G4);
  const _Float16* Wd  = w_hh + (size_t)dir * (G4 * Hsz);
  float*          hod = h_out + (size_t)dir * (Bsz * Ssz * Hsz);

  __shared__ __align__(16) _Float16 hsh[2][Bsz][HP];

  int tid  = threadIdx.x;
  int lane = tid & 31;
  int w    = tid >> 5;        // wave id 0..15
  int half = lane >> 4;
  int n    = lane & 15;
  int jcol = (w << 4) + n;    // hidden column this lane writes

  // zero init state buffer 0
  {
    _Float16* hp = &hsh[0][0][0];
    for (int i = tid; i < Bsz * HP; i += 512) hp[i] = (_Float16)0.0f;
  }
  v8f creg, hreg;
#pragma unroll
  for (int r = 0; r < 8; ++r) { creg[r] = 0.0f; hreg[r] = 0.0f; }
  __syncthreads();

  int cur = 0;
  for (int t = 0; t < Ssz; ++t) {
    int s = dir ? (Ssz - 1 - t) : t;

    // Prefetch next step's xp slice (L2 temporal) — global_prefetch_b8
    if (t < Ssz - 1) {
      int sn = dir ? (Ssz - 2 - t) : (t + 1);
#pragma unroll
      for (int g = 0; g < 4; ++g) {
        int b = (half << 3);
        __builtin_prefetch(&xpd[((size_t)b * Ssz + sn) * G4 + (g << 8) + jcol], 0, 1);
      }
    }

    // A fragments: h (shared, f16) — same for all 4 gate groups
    v16h afr[8];
    {
      int m  = lane & 15;
      int kb = (lane >> 4) << 3;
#pragma unroll
      for (int kk = 0; kk < 8; ++kk) {
        const _Float16* rp = &hsh[cur][m][kb + (kk << 5)];
        v8h lo = *(const v8h*)(rp);
        v8h hi = *(const v8h*)(rp + 16);
#pragma unroll
        for (int i = 0; i < 8; ++i) { afr[kk][i] = lo[i]; afr[kk][i + 8] = hi[i]; }
      }
    }

    // gates[b, 256*g + 16w + n] = xp[...] + h @ w_hh^T   (4 tiles, 8 WMMA each)
    v8f acc[4];
#pragma unroll
    for (int g = 0; g < 4; ++g) {
      int col = (g << 8) + (w << 4);
#pragma unroll
      for (int r = 0; r < 8; ++r) {
        int b = r + (half << 3);
        acc[g][r] = xpd[((size_t)b * Ssz + s) * G4 + col + n];
      }
      const _Float16* Bt = Wd + (size_t)col * Hsz;
#pragma unroll
      for (int kk = 0; kk < 8; ++kk)
        acc[g] = wmma_f16(afr[kk], load_frag(Bt + (kk << 5), Hsz), acc[g]);
    }

    // elementwise LSTM update with packed-sequence masking
    int nxt = cur ^ 1;
#pragma unroll
    for (int r = 0; r < 8; ++r) {
      int b = r + (half << 3);
      float iv = sigmoidf_(acc[0][r]);
      float fv = sigmoidf_(acc[1][r]);
      float gv = tanhf(acc[2][r]);
      float ov = sigmoidf_(acc[3][r]);
      float cn = fv * creg[r] + iv * gv;
      float hn = ov * tanhf(cn);
      bool  mk = mask[b * Ssz + s] != 0;
      creg[r] = mk ? cn : creg[r];
      hreg[r] = mk ? hn : hreg[r];
      hod[((size_t)b * Ssz + s) * Hsz + jcol] = mk ? hreg[r] : 0.0f;
      hsh[nxt][b][jcol] = (_Float16)hreg[r];
    }
    __syncthreads();
    cur = nxt;
  }
}

// ----------------------------------------------------------------- k_gate ---
// probs[pos] = mask ? sigmoid(h_f·z_w[0:256] + h_b·z_w[256:512] + z_b) : 0
__global__ __launch_bounds__(256) void k_gate(const float* __restrict__ h_out,
                                              const float* __restrict__ zw,
                                              const float* __restrict__ zb,
                                              const unsigned char* __restrict__ mask,
                                              float* __restrict__ probs) {
  int wid  = (blockIdx.x * blockDim.x + threadIdx.x) >> 5;   // 0..16383
  int lane = threadIdx.x & 31;
  const float* hf = h_out + (size_t)wid * Hsz;
  const float* hb = h_out + (size_t)(Bsz * Ssz) * Hsz + (size_t)wid * Hsz;
  float v = 0.0f;
  for (int i = lane; i < Hsz; i += 32) v += hf[i] * zw[i] + hb[i] * zw[Hsz + i];
#pragma unroll
  for (int off = 16; off > 0; off >>= 1) v += __shfl_down(v, off, 32);
  if (lane == 0)
    probs[wid] = mask[wid] ? sigmoidf_(v + zb[0]) : 0.0f;
}

// ----------------------------------------------------------------- k_topk ---
// z[i] = 1 iff rank(probs[i]) < round(0.1*len) and probs[i] > 0.
// Stable-argsort tie-break: higher prob first, then lower index.
__global__ __launch_bounds__(1024) void k_topk(const float* __restrict__ probs,
                                               const unsigned char* __restrict__ mask,
                                               float* __restrict__ z) {
  int b = blockIdx.x, i = threadIdx.x;
  __shared__ float p[Ssz];
  __shared__ int   lensh;
  if (i == 0) lensh = 0;
  float pi = probs[b * Ssz + i];
  p[i] = pi;
  __syncthreads();
  if (mask[b * Ssz + i]) atomicAdd(&lensh, 1);
  __syncthreads();
  int k = (int)rintf(0.10f * (float)lensh);   // RNE matches jnp.round
  int cnt = 0;
  for (int j = 0; j < Ssz; ++j) {
    float pj = p[j];
    cnt += ((pj > pi) || (pj == pi && j < i)) ? 1 : 0;
  }
  z[b * Ssz + i] = (cnt < k && pi > 0.0f) ? 1.0f : 0.0f;
}

// ---------------------------------------------------------------------------
extern "C" void kernel_launch(void* const* d_in, const int* in_sizes, int n_in,
                              void* d_out, int out_size, void* d_ws, size_t ws_size,
                              hipStream_t stream) {
  const int*           x      = (const int*)d_in[0];
  const unsigned char* mask   = (const unsigned char*)d_in[1];   // jax bool -> 1 byte
  const float*         etab   = (const float*)d_in[2];
  const float*         w_ih_f = (const float*)d_in[3];
  const float*         w_hh_f = (const float*)d_in[4];
  const float*         b_f    = (const float*)d_in[5];
  const float*         w_ih_b = (const float*)d_in[6];
  const float*         w_hh_b = (const float*)d_in[7];
  const float*         b_b    = (const float*)d_in[8];
  const float*         z_w    = (const float*)d_in[9];
  const float*         z_b    = (const float*)d_in[10];
  float*               zout   = (float*)d_out;

  // workspace carve-up (256B aligned)
  char* ws = (char*)d_ws;
  size_t off = 0;
  auto take = [&](size_t bytes) -> char* {
    char* p = ws + off;
    off = (off + bytes + 255) & ~(size_t)255;
    return p;
  };
  _Float16* emb_h = (_Float16*)take((size_t)Bsz * Ssz * Esz * 2);       //   8 MB
  _Float16* wih_h = (_Float16*)take((size_t)2 * G4 * Esz * 2);          //   1 MB
  _Float16* whh_h = (_Float16*)take((size_t)2 * G4 * Hsz * 2);          //   1 MB
  float*    xp    = (float*)take((size_t)2 * Bsz * Ssz * G4 * 4);       // 134 MB
  float*    h_out = (float*)take((size_t)2 * Bsz * Ssz * Hsz * 4);      //  34 MB
  float*    probs = (float*)take((size_t)Bsz * Ssz * 4);                //  64 KB
  (void)ws_size; (void)in_sizes; (void)n_in; (void)out_size;

  // 1) embedding gather -> f16
  k_embed<<<(Bsz * Ssz * Esz) / 256, 256, 0, stream>>>(x, etab, emb_h);

  // 2) weight conversions -> f16 (row-major [1024,256])
  const int WN = G4 * Esz;                 // 262144
  k_cvt<<<(WN + 255) / 256, 256, 0, stream>>>(w_ih_f, wih_h,       WN);
  k_cvt<<<(WN + 255) / 256, 256, 0, stream>>>(w_ih_b, wih_h + WN,  WN);
  k_cvt<<<(WN + 255) / 256, 256, 0, stream>>>(w_hh_f, whh_h,       WN);
  k_cvt<<<(WN + 255) / 256, 256, 0, stream>>>(w_hh_b, whh_h + WN,  WN);

  // 3) input projections (both directions): 131072 wave-tiles / 8 per block
  k_inproj<<<16384, 256, 0, stream>>>(emb_h, wih_h, b_f, b_b, xp);

  // 4) sequential bidirectional LSTM scan: 1 workgroup per direction
  k_lstm<<<2, 512, 0, stream>>>(xp, whh_h, mask, h_out);

  // 5) gate probabilities
  k_gate<<<(Bsz * Ssz) / 8, 256, 0, stream>>>(h_out, z_w, z_b, mask, probs);

  // 6) per-row top-k selection
  k_topk<<<Bsz, 1024, 0, stream>>>(probs, mask, zout);
}
